// BackwardConv2D_13829794693485
// MI455X (gfx1250) — matrix operand order repaired and checked by reading, not compiled, not run on hardware
//
#include <hip/hip_runtime.h>

typedef __bf16 bf16;
typedef __attribute__((ext_vector_type(16))) __bf16 v16bf;
typedef __attribute__((ext_vector_type(8)))  __bf16 v8bf;
typedef __attribute__((ext_vector_type(8)))  float  v8f;

// Problem constants (from reference)
#define BQ    8
#define HQ    32
#define WQ    32
#define CINQ  64
#define COUTQ 64
#define NOUTQ 32
#define NFLAT 65536   // H*W*COUT == H*W*CIN

__device__ __forceinline__ v16bf frag_combine(v8bf lo, v8bf hi) {
    v16bf r;
    #pragma unroll
    for (int i = 0; i < 8; ++i) { r[i] = lo[i]; r[8 + i] = hi[i]; }
    return r;
}

// Load a 16-lane-striped bf16 fragment from LDS: two contiguous 16B chunks,
// second chunk `hioff` elements after the first (A: +16 elems, B: +16 elems).
__device__ __forceinline__ v16bf frag_load(const bf16* p, int hioff) {
    return frag_combine(*(const v8bf*)p, *(const v8bf*)(p + hioff));
}

// ---------------------------------------------------------------------------
// Transposed-conv via implicit GEMM on v_wmma_f32_16x16x32_bf16.
// Block: 256 threads (8 wave32). One block = one (b, n, sel, 4-row strip).
// M = 4 rows * 32 cols = 128 positions -> 8 M-tiles of 16, one per wave.
// N = CIN = 64 -> 4 N-tiles of 16 (A-fragment reused across N-tiles).
// K = KH*KW*COUT = 576, looped as 3 kh-slices * 6 ksteps of 32.
// Inner loop is software-pipelined one kstep ahead so LDS loads overlap WMMA.
// ---------------------------------------------------------------------------
__global__ __launch_bounds__(256)
void convt_wmma_kernel(const float* __restrict__ wu,
                       const float* __restrict__ wl,
                       const float* __restrict__ kern,
                       float* __restrict__ out_u,
                       float* __restrict__ out_l)
{
    // input strip: rows h0-1 .. h0+4 (6 rows), w padded by 1 each side (34), bf16
    __shared__ __align__(16) bf16 sI[6 * 34 * 64];   // 26112 B
    // kernel slice for current kh: sK[cin][kw*64 + cout], bf16
    __shared__ __align__(16) bf16 sK[192 * 64];      // 24576 B

    const int tid = threadIdx.x;
    const int bid = blockIdx.x;            // 4096 blocks = 2 sel * 32 n * 8 b * 8 strips
    const int sel = bid & 1;
    const int n   = (bid >> 1) & 31;
    const int b   = (bid >> 6) & 7;
    const int h0  = ((bid >> 9) & 7) << 2; // 4-row strip

    const float* __restrict__ src = sel ? wl : wu;
    float* __restrict__ dst       = sel ? out_l : out_u;

    // ---- stage input strip (zero-padded halo), fp32 -> bf16 ----
    for (int idx = tid; idx < 6 * 34 * 64; idx += 256) {
        const int c  = idx & 63;
        const int wc = (idx >> 6) % 34;
        const int r  = idx / (34 * 64);
        const int hp = h0 - 1 + r;
        const int wp = wc - 1;
        float v = 0.0f;
        if (hp >= 0 && hp < HQ && wp >= 0 && wp < WQ) {
            const size_t g = (((size_t)b * NFLAT +
                               (size_t)(hp * WQ + wp) * COUTQ + c) << 5) + n;
            __builtin_prefetch(&src[g + (1 << 13)], 0, 1);  // global_prefetch_b8
            v = src[g];
        }
        sI[idx] = (bf16)v;
    }

    const int lane   = tid & 31;
    const int waveid = tid >> 5;
    const int mrow   = lane & 15;   // M index within tile (A rows / D N-group)
    const int khalf  = lane >> 4;   // K-half selector per ISA A/B layout

    v8f acc[4] = {};                // one f32 16x16 accumulator per N-tile

    // wave owns M-tile = waveid: positions p = waveid*16 .. +15 (row-major)
    const int p  = (waveid << 4) + mrow;
    const int pr = p >> 5;          // row within strip (0..3)
    const int pw = p & 31;          // col
    const int acolo = (khalf << 3); // A fragment K-half column offset

    for (int kh = 0; kh < 3; ++kh) {
        __syncthreads();
        // ---- stage kernel slice for this kh: sK[cin][kw*64+cout] ----
        for (int idx = tid; idx < 192 * 64; idx += 256) {
            const int cin  = idx & 63;
            const int kk   = idx >> 6;      // kw*64 + cout
            const int kw   = kk >> 6;
            const int cout = kk & 63;
            sK[cin * 192 + kk] = (bf16)kern[((size_t)((kh * 3 + kw) * 64 + cin) << 6) + cout];
        }
        __syncthreads();

        const int rI = pr + 2 - kh;         // sI row index (halo layout)

        // ---- software-pipelined K loop: preload ks+1 before WMMAs of ks ----
        // A addr for kstep ks: sI[(rI*34 + pw+2-(ks>>1))*64 + ((ks&1)<<5) + acolo]
        // B addr for kstep ks: sK[(nt*16 + mrow)*192 + ks*32 + khalf*8]
        v16bf aC = frag_load(&sI[(rI * 34 + pw + 2) * 64 + acolo], 16);
        v16bf bC[4];
        {
            const int kr0 = khalf << 3;
            #pragma unroll
            for (int nt = 0; nt < 4; ++nt)
                bC[nt] = frag_load(&sK[((nt << 4) + mrow) * 192 + kr0], 16);
        }

        #pragma unroll
        for (int ks = 0; ks < 6; ++ks) {
            v16bf aN = aC;
            v16bf bN[4] = { bC[0], bC[1], bC[2], bC[3] };
            if (ks < 5) {
                const int ks1 = ks + 1;
                const int kw1 = ks1 >> 1;
                const int c01 = (ks1 & 1) << 5;
                aN = frag_load(&sI[(rI * 34 + pw + 2 - kw1) * 64 + c01 + acolo], 16);
                const int kr1 = (ks1 << 5) + (khalf << 3);
                #pragma unroll
                for (int nt = 0; nt < 4; ++nt)
                    bN[nt] = frag_load(&sK[((nt << 4) + mrow) * 192 + kr1], 16);
            }
            #pragma unroll
            for (int nt = 0; nt < 4; ++nt) {
                acc[nt] = __builtin_amdgcn_wmma_f32_16x16x32_bf16(
                    false, aC, false, bC[nt], (short)0, acc[nt], false, false);
            }
            aC = aN;
            #pragma unroll
            for (int nt = 0; nt < 4; ++nt) bC[nt] = bN[nt];
        }
    }

    // ---- store: D layout: VGPR r -> M = r + 8*(lane>>4); N = lane&15 ----
    #pragma unroll
    for (int nt = 0; nt < 4; ++nt) {
        #pragma unroll
        for (int r = 0; r < 8; ++r) {
            const int M  = r + (khalf << 3);
            const int pp = (waveid << 4) + M;
            const int hh = h0 + (pp >> 5);
            const int wq = pp & 31;
            const int ci = (nt << 4) + mrow;
            dst[(((size_t)b * NFLAT +
                  (size_t)(hh * WQ + wq) * CINQ + ci) << 5) + n] = acc[nt][r];
        }
    }
}

// ---------------------------------------------------------------------------
// b_u[b,n] = sum_i w[b,0,i,n] * bias[i % 64] + b_out[b,n]   (512 reductions)
// Demonstrates gfx1250 async global->LDS staging when the builtins exist.
// ---------------------------------------------------------------------------
__global__ __launch_bounds__(256)
void bias_kernel(const float* __restrict__ wu, const float* __restrict__ wl,
                 const float* __restrict__ bu, const float* __restrict__ bl,
                 const float* __restrict__ bias,
                 float* __restrict__ out_bu, float* __restrict__ out_bl)
{
    __shared__ float sred[256];
    __shared__ __align__(16) float sbias[64];
    const int tid = threadIdx.x;
    const int bid = blockIdx.x;   // 512
    const int sel = bid & 1;
    const int n   = (bid >> 1) & 31;
    const int b   = (bid >> 6) & 7;

    const float* __restrict__ src  = sel ? wl : wu;
    const float* __restrict__ badd = sel ? bl : bu;
    float* __restrict__ dst        = sel ? out_bl : out_bu;

#if defined(__gfx1250__) &&                                            \
    __has_builtin(__builtin_amdgcn_global_load_async_to_lds_b32) &&    \
    __has_builtin(__builtin_amdgcn_s_wait_asynccnt)
    if (tid < 64) {
        __builtin_amdgcn_global_load_async_to_lds_b32(
            (__attribute__((address_space(1))) int*)(bias + tid),
            (__attribute__((address_space(3))) int*)(sbias + tid),
            0, 0);
    }
    __builtin_amdgcn_s_wait_asynccnt(0);
#else
    if (tid < 64) sbias[tid] = bias[tid];
#endif
    __syncthreads();

    float s = 0.0f;
    for (int i = tid; i < NFLAT; i += 256) {
        s += src[(((size_t)b * NFLAT + i) << 5) + n] * sbias[i & 63];
    }
    sred[tid] = s;
    __syncthreads();
    #pragma unroll
    for (int off = 128; off > 0; off >>= 1) {
        if (tid < off) sred[tid] += sred[tid + off];
        __syncthreads();
    }
    if (tid == 0) dst[b * 32 + n] = sred[0] + badd[b * 32 + n];
}

// ---------------------------------------------------------------------------
extern "C" void kernel_launch(void* const* d_in, const int* in_sizes, int n_in,
                              void* d_out, int out_size, void* d_ws, size_t ws_size,
                              hipStream_t stream)
{
    (void)in_sizes; (void)n_in; (void)d_ws; (void)ws_size; (void)out_size;
    // inputs: 0:x(unused) 1:w_out_u 2:b_out_u 3:w_out_l 4:b_out_l 5:kernel 6:bias
    const float* wu   = (const float*)d_in[1];
    const float* bu   = (const float*)d_in[2];
    const float* wl   = (const float*)d_in[3];
    const float* bl   = (const float*)d_in[4];
    const float* kern = (const float*)d_in[5];
    const float* bias = (const float*)d_in[6];

    float* out    = (float*)d_out;
    float* out_wu = out;                 // 8*1*65536*32 = 16777216
    float* out_bu = out + 16777216;      // 8*1*32       = 256
    float* out_wl = out + 16777472;      // 16777216
    float* out_bl = out + 33554688;      // 256

    convt_wmma_kernel<<<4096, 256, 0, stream>>>(wu, wl, kern, out_wu, out_wl);
    bias_kernel<<<512, 256, 0, stream>>>(wu, wl, bu, bl, bias, out_bu, out_bl);
}
